// MultiHeadAttention_335007449467
// MI455X (gfx1250) — matrix operand order repaired and checked by reading, not compiled, not run on hardware
//
#include <hip/hip_runtime.h>
#include <hip/hip_bf16.h>
#include <stdint.h>

#define D_MODEL 1024
#define NUM_HEADS 16
#define DEPTH 64
#define BB 2
#define SS 2048
#define MS (BB * SS) // 4096 flattened [B,S] rows

typedef __attribute__((ext_vector_type(16))) __bf16 v16bf;
typedef __attribute__((ext_vector_type(8)))  float  v8f;
typedef __attribute__((ext_vector_type(4)))  unsigned int u32x4;
typedef __attribute__((ext_vector_type(4)))  float  f32x4;

union FragB16 { v16bf v; u32x4 q[2]; };

__device__ __forceinline__ unsigned short f2bf(float f) {
    return __builtin_bit_cast(unsigned short, (__bf16)f);
}

// ---------------------------------------------------------------------------
// fp32 -> bf16 elementwise conversion (grid-stride)
// ---------------------------------------------------------------------------
__global__ __launch_bounds__(256) void cvt_f32_bf16(const float* __restrict__ s,
                                                    unsigned short* __restrict__ d,
                                                    int n) {
    int i = blockIdx.x * blockDim.x + threadIdx.x;
    int stride = gridDim.x * blockDim.x;
    for (; i < n; i += stride) d[i] = f2bf(s[i]);
}

// ---------------------------------------------------------------------------
// Generic Y = X * W^T (+bias) * alpha, bf16 inputs, WMMA bf16->f32 accum.
//   A: [M,K] row-major bf16 (lda)     -> A-matrix fragments
//   W: [N,K] row-major bf16 (ldw)     -> B[k,n] = W[n,k]; column n of B is a
//                                        contiguous row of W.
// Per wave: 16x64 output tile = 4 accumulators, K-steps of 32.
// z dimension: z -> (zb = z/hdiv, zh = z%hdiv) with per-operand offsets,
// used to walk (batch, head) slices for the QK^T scores GEMM.
// Outputs: fp32 (OUT_F32), plain bf16, or transposed bf16 (V -> Vt[b,h,d,s]).
// ---------------------------------------------------------------------------
template <bool OUT_F32, bool TRANS_STORE>
__global__ __launch_bounds__(256) void gemm_bf16_rr(
    const unsigned short* __restrict__ A, int lda,
    const unsigned short* __restrict__ W, int ldw,
    float* __restrict__ Cf, int ldc,
    unsigned short* __restrict__ Cb, int ldcb,
    const float* __restrict__ bias,
    float alpha, int M, int N, int K,
    int hdiv,
    long aZb, long aZh, long wZb, long wZh, long cZb, long cZh)
{
    const int z  = blockIdx.z;
    const int zb = z / hdiv;
    const int zh = z - zb * hdiv;
    A += (size_t)zb * aZb + (size_t)zh * aZh;
    W += (size_t)zb * wZb + (size_t)zh * wZh;
    const size_t coff = (size_t)zb * cZb + (size_t)zh * cZh;

    const int wave = threadIdx.x >> 5;
    const int lane = threadIdx.x & 31;
    const int l15  = lane & 15;
    const int lh   = lane >> 4;   // 0: lanes 0-15, 1: lanes 16-31

    const int m0 = blockIdx.x * 128 + wave * 16;
    const int n0 = blockIdx.y * 64;

    v8f acc[4] = {};

    // A-matrix 16x32 bf16 layout: lanes 0-15 row M=l15 hold K 0..7 & 16..23,
    // lanes 16-31 hold K 8..15 & 24..31 -> two contiguous b128 loads per lane.
    const unsigned short* Arow = A + (size_t)(m0 + l15) * lda;

    for (int k0 = 0; k0 < K; k0 += 32) {
        FragB16 a;
        const unsigned short* ap = Arow + k0 + lh * 8;
        a.q[0] = *(const u32x4*)(ap);
        a.q[1] = *(const u32x4*)(ap + 16);
        __builtin_prefetch(ap + 64, 0, 1);   // two k-steps ahead -> global_prefetch_b8
#pragma unroll
        for (int t = 0; t < 4; ++t) {
            // B-matrix 32x16: lane column N=l15; lanes 0-15 K 0..7 & 16..23,
            // lanes 16-31 K 8..15 & 24..31 -> two b128 loads from W row n.
            const unsigned short* wp =
                W + (size_t)(n0 + t * 16 + l15) * ldw + k0 + lh * 8;
            FragB16 b;
            b.q[0] = *(const u32x4*)(wp);
            b.q[1] = *(const u32x4*)(wp + 16);
            acc[t] = __builtin_amdgcn_wmma_f32_16x16x32_bf16(
                false, a.v, false, b.v, (short)0, acc[t], false, false);
        }
    }

    // D 16x16 f32 layout: lane holds column N; VGPR j is row M = j + 8*lh.
#pragma unroll
    for (int t = 0; t < 4; ++t) {
        const int n = n0 + t * 16 + l15;
        const float bv = bias ? bias[n] : 0.0f;
#pragma unroll
        for (int j = 0; j < 8; ++j) {
            const int row = m0 + lh * 8 + j;
            const float val = acc[t][j] * alpha + bv;
            if constexpr (OUT_F32) {
                Cf[coff + (size_t)row * ldc + n] = val;
            } else if constexpr (TRANS_STORE) {
                // V head-transposed store: Vt[b, h*64+d, s], S = 2048
                const int bb = row >> 11;
                const int s  = row & 2047;
                Cb[coff + (size_t)(bb * D_MODEL + n) * ldcb + s] = f2bf(val);
            } else {
                Cb[coff + (size_t)row * ldcb + n] = f2bf(val);
            }
        }
    }
}

// ---------------------------------------------------------------------------
// Row softmax, in place, one 256-thread block per row of 2048 fp32.
// ---------------------------------------------------------------------------
__global__ __launch_bounds__(256) void softmax_rows(float* __restrict__ P)
{
    float* p = P + (size_t)blockIdx.x * SS;
    const int t = threadIdx.x;

    f32x4 x0 = ((const f32x4*)p)[t * 2];
    f32x4 x1 = ((const f32x4*)p)[t * 2 + 1];

    float m = x0[0];
#pragma unroll
    for (int i = 1; i < 4; ++i) m = fmaxf(m, x0[i]);
#pragma unroll
    for (int i = 0; i < 4; ++i) m = fmaxf(m, x1[i]);
#pragma unroll
    for (int o = 16; o; o >>= 1) m = fmaxf(m, __shfl_xor(m, o, 32));

    __shared__ float smax[8];
    __shared__ float ssum[8];
    if ((t & 31) == 0) smax[t >> 5] = m;
    __syncthreads();
    float rmax = smax[0];
#pragma unroll
    for (int i = 1; i < 8; ++i) rmax = fmaxf(rmax, smax[i]);

    float s = 0.0f;
#pragma unroll
    for (int i = 0; i < 4; ++i) { x0[i] = __expf(x0[i] - rmax); s += x0[i]; }
#pragma unroll
    for (int i = 0; i < 4; ++i) { x1[i] = __expf(x1[i] - rmax); s += x1[i]; }
#pragma unroll
    for (int o = 16; o; o >>= 1) s += __shfl_xor(s, o, 32);
    if ((t & 31) == 0) ssum[t >> 5] = s;
    __syncthreads();
    float rsum = ssum[0];
#pragma unroll
    for (int i = 1; i < 8; ++i) rsum += ssum[i];

    const float inv = 1.0f / rsum;
#pragma unroll
    for (int i = 0; i < 4; ++i) x0[i] *= inv;
#pragma unroll
    for (int i = 0; i < 4; ++i) x1[i] *= inv;
    ((f32x4*)p)[t * 2]     = x0;
    ((f32x4*)p)[t * 2 + 1] = x1;
}

// ---------------------------------------------------------------------------
// attn = P @ V per (b,h): A = fp32 probs (cvt to bf16 in regs),
// B = Vt[b,h][d, s] bf16 (row-contiguous over k), out = merged-head bf16.
// M = 2048, N = 64, K = 2048.
// ---------------------------------------------------------------------------
__global__ __launch_bounds__(256) void gemm_pv(
    const float* __restrict__ P,            // [B*H][2048][2048] fp32 probs
    const unsigned short* __restrict__ Vt,  // [B*H][64][2048] bf16
    unsigned short* __restrict__ Attn)      // [4096][1024] bf16 merged heads
{
    const int z = blockIdx.z;               // b*16 + h
    const int b = z >> 4;
    const int h = z & 15;
    const float* Pz = P + (size_t)z * SS * SS;
    const unsigned short* Vz = Vt + (size_t)z * DEPTH * SS;

    const int wave = threadIdx.x >> 5;
    const int lane = threadIdx.x & 31;
    const int l15  = lane & 15;
    const int lh   = lane >> 4;
    const int m0   = blockIdx.x * 128 + wave * 16;

    v8f acc[4] = {};
    const float* Prow = Pz + (size_t)(m0 + l15) * SS;

    for (int k0 = 0; k0 < SS; k0 += 32) {
        float tf[16];
        const int ka = k0 + lh * 8;
        ((f32x4*)tf)[0] = *(const f32x4*)(Prow + ka);
        ((f32x4*)tf)[1] = *(const f32x4*)(Prow + ka + 4);
        ((f32x4*)tf)[2] = *(const f32x4*)(Prow + ka + 16);
        ((f32x4*)tf)[3] = *(const f32x4*)(Prow + ka + 20);
        FragB16 a;
#pragma unroll
        for (int i = 0; i < 16; ++i) a.v[i] = (__bf16)tf[i];
        __builtin_prefetch(Prow + ka + 64, 0, 1);
#pragma unroll
        for (int t = 0; t < 4; ++t) {
            const unsigned short* wp =
                Vz + (size_t)(t * 16 + l15) * SS + k0 + lh * 8;
            FragB16 bf;
            bf.q[0] = *(const u32x4*)(wp);
            bf.q[1] = *(const u32x4*)(wp + 16);
            acc[t] = __builtin_amdgcn_wmma_f32_16x16x32_bf16(
                false, a.v, false, bf.v, (short)0, acc[t], false, false);
        }
    }

#pragma unroll
    for (int t = 0; t < 4; ++t) {
        const int d = t * 16 + l15;
#pragma unroll
        for (int j = 0; j < 8; ++j) {
            const int srow = m0 + lh * 8 + j;
            Attn[(size_t)(b * SS + srow) * D_MODEL + h * DEPTH + d] =
                f2bf(acc[t][j]);
        }
    }
}

// ---------------------------------------------------------------------------
extern "C" void kernel_launch(void* const* d_in, const int* in_sizes, int n_in,
                              void* d_out, int out_size, void* d_ws, size_t ws_size,
                              hipStream_t stream)
{
    const float* q    = (const float*)d_in[0];
    const float* k    = (const float*)d_in[1];
    const float* v    = (const float*)d_in[2];
    const float* wq_w = (const float*)d_in[3];
    const float* wq_b = (const float*)d_in[4];
    const float* wk_w = (const float*)d_in[5];
    const float* wk_b = (const float*)d_in[6];
    const float* wv_w = (const float*)d_in[7];
    const float* wv_b = (const float*)d_in[8];
    const float* fc_w = (const float*)d_in[9];
    const float* fc_b = (const float*)d_in[10];

    float* out   = (float*)d_out;                     // [4096,1024]
    float* attnw = out + (size_t)MS * D_MODEL;        // [2,16,2048,2048]

    unsigned short* ws = (unsigned short*)d_ws;
    size_t o = 0;
    unsigned short* qb  = ws + o; o += (size_t)MS * D_MODEL;
    unsigned short* kb  = ws + o; o += (size_t)MS * D_MODEL;
    unsigned short* vb  = ws + o; o += (size_t)MS * D_MODEL;
    unsigned short* wqb = ws + o; o += (size_t)D_MODEL * D_MODEL;
    unsigned short* wkb = ws + o; o += (size_t)D_MODEL * D_MODEL;
    unsigned short* wvb = ws + o; o += (size_t)D_MODEL * D_MODEL;
    unsigned short* fcb = ws + o; o += (size_t)D_MODEL * D_MODEL;
    unsigned short* Qh  = ws + o; o += (size_t)MS * D_MODEL;       // proj Q bf16
    unsigned short* Kh  = ws + o; o += (size_t)MS * D_MODEL;       // proj K bf16
    unsigned short* Vt  = ws + o; o += (size_t)BB * D_MODEL * SS;  // V transposed
    unsigned short* Ab  = ws + o; o += (size_t)MS * D_MODEL;       // merged attn

    const dim3 blk(256);

    // 1) fp32 -> bf16 conversions
    cvt_f32_bf16<<<dim3(4096), blk, 0, stream>>>(q, qb, MS * D_MODEL);
    cvt_f32_bf16<<<dim3(4096), blk, 0, stream>>>(k, kb, MS * D_MODEL);
    cvt_f32_bf16<<<dim3(4096), blk, 0, stream>>>(v, vb, MS * D_MODEL);
    cvt_f32_bf16<<<dim3(1024), blk, 0, stream>>>(wq_w, wqb, D_MODEL * D_MODEL);
    cvt_f32_bf16<<<dim3(1024), blk, 0, stream>>>(wk_w, wkb, D_MODEL * D_MODEL);
    cvt_f32_bf16<<<dim3(1024), blk, 0, stream>>>(wv_w, wvb, D_MODEL * D_MODEL);
    cvt_f32_bf16<<<dim3(1024), blk, 0, stream>>>(fc_w, fcb, D_MODEL * D_MODEL);

    // 2) Q/K/V projections (y = x W^T + b)
    const dim3 gproj(MS / 128, D_MODEL / 64, 1);
    gemm_bf16_rr<false, false><<<gproj, blk, 0, stream>>>(
        qb, D_MODEL, wqb, D_MODEL, nullptr, 0, Qh, D_MODEL, wq_b, 1.0f,
        MS, D_MODEL, D_MODEL, 1, 0, 0, 0, 0, 0, 0);
    gemm_bf16_rr<false, false><<<gproj, blk, 0, stream>>>(
        kb, D_MODEL, wkb, D_MODEL, nullptr, 0, Kh, D_MODEL, wk_b, 1.0f,
        MS, D_MODEL, D_MODEL, 1, 0, 0, 0, 0, 0, 0);
    gemm_bf16_rr<false, true><<<gproj, blk, 0, stream>>>(
        vb, D_MODEL, wvb, D_MODEL, nullptr, 0, Vt, SS, wv_b, 1.0f,
        MS, D_MODEL, D_MODEL, 1, 0, 0, 0, 0, 0, 0);

    // 3) scores = (Qh Kh^T) / sqrt(64): per (b,h) slice, "W" = K head rows
    const long sBZ = (long)SS * D_MODEL;   // batch stride into Qh/Kh
    const long sHZ = (long)DEPTH;          // head column offset
    const long cBZ = (long)NUM_HEADS * SS * SS;
    const long cHZ = (long)SS * SS;
    gemm_bf16_rr<true, false><<<dim3(SS / 128, SS / 64, BB * NUM_HEADS),
                                blk, 0, stream>>>(
        Qh, D_MODEL, Kh, D_MODEL, attnw, SS, nullptr, 0, nullptr, 0.125f,
        SS, SS, DEPTH, NUM_HEADS, sBZ, sHZ, sBZ, sHZ, cBZ, cHZ);

    // 4) softmax over each of the B*H*S rows, in place in d_out
    softmax_rows<<<dim3(BB * NUM_HEADS * SS), blk, 0, stream>>>(attnw);

    // 5) attn = P @ V (merged-head bf16 output)
    gemm_pv<<<dim3(SS / 128, 1, BB * NUM_HEADS), blk, 0, stream>>>(attnw, Vt, Ab);

    // 6) out = attn fc^T + fc_b (fp32)
    gemm_bf16_rr<true, false><<<gproj, blk, 0, stream>>>(
        Ab, D_MODEL, fcb, D_MODEL, out, D_MODEL, nullptr, 0, fc_b, 1.0f,
        MS, D_MODEL, D_MODEL, 1, 0, 0, 0, 0, 0, 0);
}
